// SingleHead_403726925840
// MI455X (gfx1250) — compile-verified, matrix operand order checked by reading, not run on hardware
//
#include <hip/hip_runtime.h>
#include <hip/hip_bf16.h>

// Problem constants (from reference): B=4, T=4096, C=1024, H=64
#define BATCH 4
#define SEQ   4096
#define CDIM  1024
#define HDIM  64

typedef __attribute__((ext_vector_type(16))) _Float16 v16h;
typedef __attribute__((ext_vector_type(8)))  _Float16 v8h;
typedef __attribute__((ext_vector_type(4)))  _Float16 v4h;
typedef __attribute__((ext_vector_type(8)))  float    v8f;
typedef __attribute__((ext_vector_type(4)))  int      v4i;

union frag16 { v16h f; v8h h[2]; };

#if __has_builtin(__builtin_amdgcn_global_load_async_to_lds_b128) && \
    __has_builtin(__builtin_amdgcn_s_wait_asynccnt)
#define HAVE_ASYNC_LDS 1
#endif

#define AS_GLOBAL __attribute__((address_space(1)))
#define AS_LOCAL  __attribute__((address_space(3)))

// Cooperative 128-bit tile copy global->LDS. Uses the CDNA5 async-LDS path
// (ASYNCcnt-tracked, no VGPR round trip) when the toolchain exposes it.
__device__ __forceinline__ void copy_tile_b128(const uint4* __restrict__ src,
                                               uint4* __restrict__ dst,
                                               int tid, int n, int stride) {
#ifdef HAVE_ASYNC_LDS
    for (int i = tid; i < n; i += stride)
        __builtin_amdgcn_global_load_async_to_lds_b128(
            (AS_GLOBAL v4i*)(src + i), (AS_LOCAL v4i*)(dst + i), 0, 0);
#else
    for (int i = tid; i < n; i += stride) dst[i] = src[i];
#endif
}

__device__ __forceinline__ void wait_async_lds() {
#ifdef HAVE_ASYNC_LDS
    __builtin_amdgcn_s_wait_asynccnt(0);
#endif
}

// Per CDNA5 ISA 7.12.2 (16-bit A/B fragments): lane L holds two contiguous
// 8-element K-runs starting at K = (L>>4)*8 and K = (L>>4)*8 + 16.
__device__ __forceinline__ int frag_kbase(int lane) { return (lane >> 4) << 3; }

// Load a 16x32-K fragment whose per-lane data is contiguous along K in LDS.
__device__ __forceinline__ v16h load_frag(const _Float16* p) {
    frag16 u;
    u.h[0] = *(const v8h*)(p);
    u.h[1] = *(const v8h*)(p + 16);
    return u.f;
}

// ---------------------------------------------------------------------------
// Kernel 1: out[M,64] = x[M,1024] @ W[1024,64]  (f32 in, f16 out, scale folded)
// Block: 256 threads = 8 waves; each wave: one 16-row M-tile x 4 N-tiles.
// LDS: x tile K-major [128][32]; W tile TRANSPOSED [n=64][k=32] so B-fragment
// gathers are contiguous -> ds_load_b128.
// ---------------------------------------------------------------------------
__global__ void __launch_bounds__(256) qkv_proj_kernel(
    const float* __restrict__ x, const float* __restrict__ W,
    _Float16* __restrict__ outp, float out_scale)
{
    __shared__ __align__(16) _Float16 ldsX[128 * 32];
    __shared__ __align__(16) _Float16 ldsWt[64 * 32];   // [n][k]

    const int tid  = threadIdx.x;
    const int lane = tid & 31;
    const int wave = tid >> 5;
    const int m0   = blockIdx.x * 128;
    const int ncol = lane & 15;
    const int kb   = frag_kbase(lane);

    v8f acc[4];
    for (int nt = 0; nt < 4; ++nt)
        for (int r = 0; r < 8; ++r) acc[nt][r] = 0.0f;

    for (int k0 = 0; k0 < CDIM; k0 += 32) {
        // Prefetch next x tile (one cacheline-ish probe per thread)
        if (k0 + 32 < CDIM)
            __builtin_prefetch(x + (size_t)(m0 + (tid >> 1)) * CDIM + k0 + 32 +
                               (tid & 1) * 16, 0, 1);
        // Stage x tile [128 x 32] f32->f16, packed 64-bit LDS stores
        {
            int basei = tid * 16;
            for (int c = 0; c < 4; ++c) {
                int idx = basei + c * 4;
                int row = idx >> 5, col = idx & 31;
                const float4 f = *(const float4*)(x + (size_t)(m0 + row) * CDIM + k0 + col);
                v4h h;
                h[0] = (_Float16)f.x; h[1] = (_Float16)f.y;
                h[2] = (_Float16)f.z; h[3] = (_Float16)f.w;
                *(v4h*)(&ldsX[row * 32 + col]) = h;
            }
        }
        // Stage W tile [32 x 64] -> transposed LDS [64][32]
        {
            int basei = tid * 8;
            for (int c = 0; c < 2; ++c) {
                int idx = basei + c * 4;
                int row = idx >> 6, col = idx & 63;
                const float4 f = *(const float4*)(W + (size_t)(k0 + row) * HDIM + col);
                ldsWt[(col + 0) * 32 + row] = (_Float16)f.x;
                ldsWt[(col + 1) * 32 + row] = (_Float16)f.y;
                ldsWt[(col + 2) * 32 + row] = (_Float16)f.z;
                ldsWt[(col + 3) * 32 + row] = (_Float16)f.w;
            }
        }
        __syncthreads();

        const int m = wave * 16 + (lane & 15);
        const v16h a = load_frag(&ldsX[m * 32 + kb]);
        for (int nt = 0; nt < 4; ++nt) {
            const v16h bfrag = load_frag(&ldsWt[(nt * 16 + ncol) * 32 + kb]);
            acc[nt] = __builtin_amdgcn_wmma_f32_16x16x32_f16(
                false, a, false, bfrag, (short)0, acc[nt], false, false);
        }
        __syncthreads();
    }

    // Store f16 result (C/D layout: lane holds rows ((lane>>4)<<3)+r, col lane&15)
    const int mbase = m0 + wave * 16 + ((lane >> 4) << 3);
    for (int nt = 0; nt < 4; ++nt)
        for (int r = 0; r < 8; ++r)
            outp[(size_t)(mbase + r) * HDIM + nt * 16 + ncol] =
                (_Float16)(acc[nt][r] * out_scale);
}

// ---------------------------------------------------------------------------
// Kernel 2: causal flash attention. Block = 128 threads (4 waves), 64 q-rows
// per block (16 per wave). K/V staged 32 rows at a time in LDS.
// LDS layouts chosen so all WMMA fragment gathers are 2x ds_load_b128:
//   qtile [m][c] (K=c contiguous), ktile [s][c] (K=c contiguous),
//   vtileT [h][s] (K=s contiguous, transposed at staging), pbuf [m][s].
// Q/K tiles staged with async global->LDS loads when available.
// q already carries the 1/sqrt(H) scale.
// ---------------------------------------------------------------------------
__global__ void __launch_bounds__(128) attn_kernel(
    const _Float16* __restrict__ q, const _Float16* __restrict__ k,
    const _Float16* __restrict__ v, float* __restrict__ outp)
{
    __shared__ __align__(16) _Float16 qtile[64 * 64];
    __shared__ __align__(16) _Float16 ktile[32 * 64];   // [s][c]
    __shared__ __align__(16) _Float16 vtileT[64 * 32];  // [h][s]
    __shared__ __align__(16) _Float16 pbuf[4][16 * 32]; // [m][s] per wave

    const int tid  = threadIdx.x;
    const int lane = tid & 31;
    const int wave = tid >> 5;
    const int b    = blockIdx.y;
    const int q0   = blockIdx.x * 64;
    const int ncol = lane & 15;
    const int kb   = frag_kbase(lane);
    const size_t base = (size_t)b * SEQ * HDIM;

    // Stage Q tile (contiguous 8KB) via async LDS copy, gather A-fragments.
    copy_tile_b128((const uint4*)(q + base + (size_t)q0 * HDIM),
                   (uint4*)qtile, tid, 512, 128);
    wait_async_lds();
    __syncthreads();

    v16h qa[2];
    {
        const int m = wave * 16 + (lane & 15);
        for (int kk = 0; kk < 2; ++kk)
            qa[kk] = load_frag(&qtile[m * 64 + kk * 32 + kb]);
    }

    v8f o[4];
    float mrow[8], lrow[8];
    for (int nt = 0; nt < 4; ++nt)
        for (int r = 0; r < 8; ++r) o[nt][r] = 0.0f;
    for (int r = 0; r < 8; ++r) { mrow[r] = -1e30f; lrow[r] = 0.0f; }

    const int qrow_base = q0 + wave * 16 + ((lane >> 4) << 3);  // +r
    const int s_end = q0 + 64;

    for (int s0 = 0; s0 < s_end; s0 += 32) {
        // Stage K tile [32 x 64] (contiguous 4KB) via async LDS copy
        copy_tile_b128((const uint4*)(k + base + (size_t)s0 * HDIM),
                       (uint4*)ktile, tid, 256, 128);
        // Prefetch next step's K/V rows while staging this one
        if (s0 + 32 < s_end) {
            __builtin_prefetch(k + base + (size_t)(s0 + 32) * HDIM + tid * 16, 0, 1);
            __builtin_prefetch(v + base + (size_t)(s0 + 32) * HDIM + tid * 16, 0, 1);
        }
        // Stage V tile transposed: global [s][h] -> LDS [h][s]
        {
            const _Float16* vsrc = v + base + (size_t)s0 * HDIM;
            int ebase = tid * 16;
            for (int c = 0; c < 2; ++c) {
                int idx  = ebase + c * 8;
                int srow = idx >> 6, hcol = idx & 63;
                v8h val = *(const v8h*)(vsrc + srow * 64 + hcol);
                for (int j = 0; j < 8; ++j)
                    vtileT[(hcol + j) * 32 + srow] = val[j];
            }
        }
        wait_async_lds();
        __syncthreads();

        // S[16 x 32] = Q[16 x 64] @ K^T[64 x 32], two 16x16 n-tiles
        v8f s[2];
        for (int j = 0; j < 2; ++j) {
            v8f c;
            for (int r = 0; r < 8; ++r) c[r] = 0.0f;
            for (int kk = 0; kk < 2; ++kk) {
                const int srow = j * 16 + ncol;  // key index within this step
                const v16h bfrag = load_frag(&ktile[srow * 64 + kk * 32 + kb]);
                c = __builtin_amdgcn_wmma_f32_16x16x32_f16(
                    false, qa[kk], false, bfrag, (short)0, c, false, false);
            }
            s[j] = c;
        }

        // Causal mask + online softmax (rows owned by this lane: qrow_base+r)
        float alpha[8];
        for (int r = 0; r < 8; ++r) {
            int qrow = qrow_base + r;
            float s0v = s[0][r], s1v = s[1][r];
            if (s0 + ncol > qrow)      s0v = -1e30f;
            if (s0 + 16 + ncol > qrow) s1v = -1e30f;
            float t = fmaxf(s0v, s1v);
            for (int off = 1; off < 16; off <<= 1)
                t = fmaxf(t, __shfl_xor(t, off, 32));
            float mnew = fmaxf(mrow[r], t);
            alpha[r] = __expf(mrow[r] - mnew);
            mrow[r] = mnew;
            float p0 = __expf(s0v - mnew);
            float p1 = __expf(s1v - mnew);
            s[0][r] = p0; s[1][r] = p1;
            lrow[r] = lrow[r] * alpha[r] + p0 + p1;
        }

        // C/D layout -> A-fragment layout via per-wave LDS buffer
        for (int r = 0; r < 8; ++r) {
            int m = ((lane >> 4) << 3) + r;
            pbuf[wave][m * 32 + ncol]      = (_Float16)s[0][r];
            pbuf[wave][m * 32 + 16 + ncol] = (_Float16)s[1][r];
        }
        const v16h pa = load_frag(&pbuf[wave][(lane & 15) * 32 + kb]);

        // O = alpha*O + P @ V   (V as [s][h] -> B fragments from vtileT[h][s])
        for (int nt = 0; nt < 4; ++nt) {
            const v16h vfrag = load_frag(&vtileT[(nt * 16 + ncol) * 32 + kb]);
            v8f oc = o[nt];
            for (int r = 0; r < 8; ++r) oc[r] *= alpha[r];
            o[nt] = __builtin_amdgcn_wmma_f32_16x16x32_f16(
                false, pa, false, vfrag, (short)0, oc, false, false);
        }
        __syncthreads();
    }

    // Finalize: divide by row sums (reduced across the 16 column lanes)
    for (int r = 0; r < 8; ++r) {
        float lsum = lrow[r];
        for (int off = 1; off < 16; off <<= 1)
            lsum += __shfl_xor(lsum, off, 32);
        float inv = 1.0f / lsum;
        int row = qrow_base + r;
        for (int nt = 0; nt < 4; ++nt)
            outp[((size_t)b * SEQ + row) * HDIM + nt * 16 + ncol] = o[nt][r] * inv;
    }
}

// ---------------------------------------------------------------------------
extern "C" void kernel_launch(void* const* d_in, const int* in_sizes, int n_in,
                              void* d_out, int out_size, void* d_ws, size_t ws_size,
                              hipStream_t stream) {
    const float* x  = (const float*)d_in[0];
    const float* Wq = (const float*)d_in[1];
    const float* Wk = (const float*)d_in[2];
    const float* Wv = (const float*)d_in[3];
    float* out = (float*)d_out;

    const size_t rows = (size_t)BATCH * SEQ;           // 16384
    _Float16* qb = (_Float16*)d_ws;
    _Float16* kb = qb + rows * HDIM;
    _Float16* vb = kb + rows * HDIM;

    const float scale = 0.125f;  // 1/sqrt(64), folded into q projection

    dim3 gp(rows / 128);  // 128 blocks
    qkv_proj_kernel<<<gp, 256, 0, stream>>>(x, Wq, qb, scale);
    qkv_proj_kernel<<<gp, 256, 0, stream>>>(x, Wk, kb, 1.0f);
    qkv_proj_kernel<<<gp, 256, 0, stream>>>(x, Wv, vb, 1.0f);

    dim3 ga(SEQ / 64, BATCH);  // 64 x 4 blocks
    attn_kernel<<<ga, 128, 0, stream>>>(qb, kb, vb, out);
}
